// CLinear_54065048322420
// MI455X (gfx1250) — matrix operand order repaired and checked by reading, not compiled, not run on hardware
//
#include <hip/hip_runtime.h>
#include <hip/hip_bf16.h>
#include <stdint.h>

typedef __attribute__((ext_vector_type(16))) _Float16 v16h;
typedef __attribute__((ext_vector_type(8)))  _Float16 v8h;
typedef __attribute__((ext_vector_type(4)))  _Float16 v4h;
typedef __attribute__((ext_vector_type(2)))  _Float16 v2h;
typedef __attribute__((ext_vector_type(8)))  float    v8f;

union FragU { v16h v; v8h h[2]; };
union Pack8 { v8h v; v2h p[4]; };
union U32H2 { uint32_t u; v2h h; };

#define KDIM 4096
#define NDIM 4096
#define MDIM 8192
#define BM 128
#define BN 128
#define BK 64
#define LDSW 72               // 64 halves + 8 pad -> 144B row stride (16B aligned)
#define NSTAGES (KDIM / BK)   // 64

__global__ __launch_bounds__(256) void clinear_wmma_f16(
    const float* __restrict__ X,      // [M, K] f32
    const int8_t* __restrict__ Q,     // [N, K] int8
    const float* __restrict__ S,      // [N, 16] f32 (scale; w = q / s)
    const float* __restrict__ Bv,     // [N] f32 bias
    float* __restrict__ Y)            // [M, N] f32
{
    // ping-pong LDS: one barrier per k-stage
    __shared__ alignas(16) _Float16 As[2][BM * LDSW];   // 2 x 18 KB
    __shared__ alignas(16) _Float16 Bs[2][BN * LDSW];   // 2 x 18 KB

    const int tid  = threadIdx.x;
    const int lane = tid & 31;
    const int wave = tid >> 5;
    const int waveM = wave & 3;   // 0..3 -> 32-row strip
    const int waveN = wave >> 2;  // 0..1 -> 64-col strip

    const int m0 = blockIdx.y * BM;
    const int n0 = blockIdx.x * BN;

    // ---- staging thread mapping ----
    const int ar = tid >> 4;              // 0..15  (A row base; +16*i)
    const int ac = (tid & 15) << 2;       // 0..60  (A float col, float4)
    const int br = tid >> 2;              // 0..63  (B row base; +64*i)
    const int bc = (tid & 3) << 4;        // 0..48  (B byte col, 16B chunk)

    v8f acc[2][4] = {};                   // 2 (M) x 4 (N) 16x16 f32 tiles

    float4 aReg[8];
    int4   bReg[2];
    float  invs[2];

    auto fetch = [&](int k0) {
        const float* xp = X + (size_t)(m0 + ar) * KDIM + k0 + ac;
        #pragma unroll
        for (int i = 0; i < 8; ++i)
            aReg[i] = *(const float4*)(xp + (size_t)(16 * i) * KDIM);
        #pragma unroll
        for (int i = 0; i < 2; ++i) {
            const int o = n0 + br + 64 * i;
            bReg[i] = *(const int4*)(Q + (size_t)o * KDIM + k0 + bc);
            // fast reciprocal (v_rcp_f32, ~1 ulp) -- result is rounded to f16 anyway
            invs[i] = __builtin_amdgcn_rcpf(S[o * 16 + ((k0 + bc) >> 8)]);
        }
    };

    auto stageStore = [&](int buf) {
        // x: f32 -> f16 (v_cvt_pk_f16_f32 pairs)
        #pragma unroll
        for (int i = 0; i < 8; ++i) {
            const float4 f = aReg[i];
            v4h hv = { (_Float16)f.x, (_Float16)f.y, (_Float16)f.z, (_Float16)f.w };
            *(v4h*)&As[buf][(size_t)(ar + 16 * i) * LDSW + ac] = hv;
        }
        // weights: int8 -> f16 via byte-magic (v_xor + 2x v_perm_b32 + packed f16 math)
        // f16(0x6400 | (q^0x80)) == 1152 + q exactly; (val - 1152) is exact (Sterbenz)
        const v2h k1152 = { (_Float16)1152.0f, (_Float16)1152.0f };
        #pragma unroll
        for (int i = 0; i < 2; ++i) {
            const _Float16 ih = (_Float16)invs[i];
            const v2h inv2 = { ih, ih };
            const uint32_t wv[4] = {
                (uint32_t)bReg[i].x ^ 0x80808080u, (uint32_t)bReg[i].y ^ 0x80808080u,
                (uint32_t)bReg[i].z ^ 0x80808080u, (uint32_t)bReg[i].w ^ 0x80808080u };
            Pack8 lo, hi;
            #pragma unroll
            for (int d = 0; d < 4; ++d) {
                U32H2 l, h;
                l.u = __builtin_amdgcn_perm(0x64646464u, wv[d], 0x05010400u); // b0,b1
                h.u = __builtin_amdgcn_perm(0x64646464u, wv[d], 0x05030402u); // b2,b3
                const v2h q0 = (l.h - k1152) * inv2;
                const v2h q1 = (h.h - k1152) * inv2;
                if (d < 2) { lo.p[2 * d] = q0; lo.p[2 * d + 1] = q1; }
                else       { hi.p[2 * (d - 2)] = q0; hi.p[2 * (d - 2) + 1] = q1; }
            }
            _Float16* dst = &Bs[buf][(size_t)(br + 64 * i) * LDSW + bc];
            *(v8h*)dst       = lo.v;
            *(v8h*)(dst + 8) = hi.v;
        }
    };

    auto compute = [&](int buf) {
        const _Float16* Abase = &As[buf][0] + (size_t)(waveM * 32 + (lane & 15)) * LDSW;
        const _Float16* Bbase = &Bs[buf][0] + (size_t)(waveN * 64 + (lane & 15)) * LDSW;
        const int aOff = 8  * (lane >> 4);   // halves
        const int bOff = 8  * (lane >> 4);   // halves
        #pragma unroll
        for (int kk = 0; kk < BK; kk += 32) {
            FragU a[2], b[4];
            #pragma unroll
            for (int mt = 0; mt < 2; ++mt) {
                const _Float16* p = Abase + (size_t)(mt * 16) * LDSW + kk + aOff;
                a[mt].h[0] = *(const v8h*)(p);
                a[mt].h[1] = *(const v8h*)(p + 16);
            }
            #pragma unroll
            for (int nt = 0; nt < 4; ++nt) {
                const _Float16* p = Bbase + (size_t)(nt * 16) * LDSW + kk + bOff;
                b[nt].h[0] = *(const v8h*)(p);
                b[nt].h[1] = *(const v8h*)(p + 16);
            }
            #pragma unroll
            for (int mt = 0; mt < 2; ++mt)
                #pragma unroll
                for (int nt = 0; nt < 4; ++nt)
                    acc[mt][nt] = __builtin_amdgcn_wmma_f32_16x16x32_f16(
                        false, a[mt].v, false, b[nt].v,
                        (short)0, acc[mt][nt], false, false);
        }
    };

    // ---------------- pipeline prologue ----------------
    fetch(0);
    stageStore(0);
    fetch(BK);
    __syncthreads();

    // ---------------- main K pipeline: one barrier per stage ----------------
    for (int s = 0; s < NSTAGES; ++s) {
        const int buf = s & 1;
        if (s + 1 < NSTAGES) stageStore(buf ^ 1);    // regs(stage s+1) -> other buffer
        if (s + 2 < NSTAGES) fetch((s + 2) * BK);    // global -> regs (overlaps WMMA)
        if (s + 3 < NSTAGES) {                       // gfx1250 global_prefetch_b8
            const int kp = (s + 3) * BK;
            __builtin_prefetch(X + (size_t)(m0 + ar) * KDIM + kp + ac, 0, 0);
            __builtin_prefetch(Q + (size_t)(n0 + br) * KDIM + kp + bc, 0, 0);
        }
        compute(buf);
        __syncthreads();
    }

    // ---------------- epilogue: bias + store ----------------
    // C/D layout: VGPR r -> (lanes 0-15: M=r, N=lane) (lanes 16-31: M=8+r, N=lane-16)
    #pragma unroll
    for (int mt = 0; mt < 2; ++mt) {
        #pragma unroll
        for (int nt = 0; nt < 4; ++nt) {
            const int n  = n0 + waveN * 64 + nt * 16 + (lane & 15);
            const float bias = Bv[n];
            const int mb = m0 + waveM * 32 + mt * 16 + ((lane >> 4) << 3);
            float* yp = Y + (size_t)mb * NDIM + n;
            #pragma unroll
            for (int r = 0; r < 8; ++r)
                yp[(size_t)r * NDIM] = acc[mt][nt][r] + bias;
        }
    }
}

extern "C" void kernel_launch(void* const* d_in, const int* in_sizes, int n_in,
                              void* d_out, int out_size, void* d_ws, size_t ws_size,
                              hipStream_t stream) {
    (void)in_sizes; (void)n_in; (void)out_size; (void)d_ws; (void)ws_size;
    const float*  x  = (const float*)d_in[0];   // [4,2048,4096] f32
    const int8_t* q  = (const int8_t*)d_in[1];  // [4096,16,256] int8
    const float*  sc = (const float*)d_in[2];   // [4096,16,1] f32
    const float*  bi = (const float*)d_in[3];   // [4096] f32
    float* y = (float*)d_out;                   // [4,2048,4096] f32

    dim3 grid(NDIM / BN, MDIM / BM);            // (32, 64)
    clinear_wmma_f16<<<grid, 256, 0, stream>>>(x, q, sc, bi, y);
}
